// AdaptiveSoftmax_60138132078906
// MI455X (gfx1250) — matrix operand order repaired and checked by reading, not compiled, not run on hardware
//
#include <hip/hip_runtime.h>
#include <hip/hip_bf16.h>

typedef __attribute__((ext_vector_type(16))) __bf16 v16bf;
typedef __attribute__((ext_vector_type(8)))  __bf16 v8bf;
typedef __attribute__((ext_vector_type(8)))  float  v8f;

#define N_TOKENS 8192
#define IN_DIM   1024

// ---------------------------------------------------------------------------
// WMMA fragment loaders from bf16 memory (16x16x32 bf16, wave32; contiguous k order, same for A and B)
// A (16x32, MxK): lane m = lane&15; K run = 16*(lane>>4)+0..15 (two 16B loads)
// B (32x16, KxN): lane n = lane&15; K run = 16*(lane>>4)+0..15 (one 32B load)
// ---------------------------------------------------------------------------
static __device__ __forceinline__ v16bf load_a_frag(const __bf16* __restrict__ A,
                                                    int lda, int row0, int k0, int lane) {
  int r    = row0 + (lane & 15);
  int koff = k0 + ((lane >> 4) << 4);
  const __bf16* p = A + (size_t)r * lda + koff;
  v8bf lo = *(const v8bf*)p;          // 16B aligned
  v8bf hi = *(const v8bf*)(p + 8);    // 16B aligned
  v16bf a;
#pragma unroll
  for (int j = 0; j < 8; ++j) { a[j] = lo[j]; a[j + 8] = hi[j]; }
  return a;
}

static __device__ __forceinline__ v16bf load_b_frag(const __bf16* __restrict__ W,
                                                    int ldw, int col0, int k0, int lane) {
  int c  = col0 + (lane & 15);
  int kb = k0 + ((lane >> 4) << 4);
  return *(const v16bf*)(W + (size_t)c * ldw + kb);   // 32B aligned
}

static __device__ __forceinline__ v8f wmma_bf16(v16bf a, v16bf b, v8f c) {
  return __builtin_amdgcn_wmma_f32_16x16x32_bf16(false, a, false, b,
                                                 (short)0, c, false, false);
}

// ---------------------------------------------------------------------------
// Kernel 0: f32 -> bf16 convert with zero tail padding (weights padded to
// their classblock span so the GEMM loops need no bounds checks).
// ---------------------------------------------------------------------------
__global__ void cvt_pad_kernel(const float* __restrict__ in, __bf16* __restrict__ out,
                               long n, long npad) {
  long stride = (long)gridDim.x * blockDim.x;
  for (long i = (long)blockIdx.x * blockDim.x + threadIdx.x; i < npad; i += stride)
    out[i] = (i < n) ? (__bf16)in[i] : (__bf16)0.0f;
}

// ---------------------------------------------------------------------------
// Kernel 1: hidden(bf16) = x @ projT   (M=8192, N=PD, K=1024)
// block = 8 waves (4 row-groups x 2 col-groups); wave tile 32x64 (2x4 WMMA);
// block tile 128 x 128.
// ---------------------------------------------------------------------------
__global__ void __launch_bounds__(256) proj_gemm_kernel(
    const __bf16* __restrict__ X, const __bf16* __restrict__ P,
    __bf16* __restrict__ H, int K, int PD) {
  int lane = threadIdx.x & 31;
  int wave = threadIdx.x >> 5;
  int row0 = blockIdx.y * 128 + (wave >> 1) * 32;
  int col0 = blockIdx.x * 128 + (wave & 1) * 64;

  v8f acc[2][4] = {};
  for (int k0 = 0; k0 < K; k0 += 32) {
    v16bf a0 = load_a_frag(X, K, row0,      k0, lane);
    v16bf a1 = load_a_frag(X, K, row0 + 16, k0, lane);
    v16bf b0 = load_b_frag(P, K, col0,      k0, lane);
    v16bf b1 = load_b_frag(P, K, col0 + 16, k0, lane);
    v16bf b2 = load_b_frag(P, K, col0 + 32, k0, lane);
    v16bf b3 = load_b_frag(P, K, col0 + 48, k0, lane);
    acc[0][0] = wmma_bf16(a0, b0, acc[0][0]);
    acc[0][1] = wmma_bf16(a0, b1, acc[0][1]);
    acc[0][2] = wmma_bf16(a0, b2, acc[0][2]);
    acc[0][3] = wmma_bf16(a0, b3, acc[0][3]);
    acc[1][0] = wmma_bf16(a1, b0, acc[1][0]);
    acc[1][1] = wmma_bf16(a1, b1, acc[1][1]);
    acc[1][2] = wmma_bf16(a1, b2, acc[1][2]);
    acc[1][3] = wmma_bf16(a1, b3, acc[1][3]);
  }
  int cg = lane >> 4, cl = lane & 15;
#pragma unroll
  for (int tr = 0; tr < 2; ++tr)
#pragma unroll
    for (int tc = 0; tc < 4; ++tc)
#pragma unroll
      for (int j = 0; j < 8; ++j) {
        int r = row0 + tr * 16 + j + 8 * cg;
        int c = col0 + tc * 16 + cl;
        H[(size_t)r * PD + c] = (__bf16)acc[tr][tc][j];
      }
}

// ---------------------------------------------------------------------------
// Kernel 2: fused logits GEMM + online logsumexp + target-logit gather.
// Block: 128 rows x 512 classes (4 chunks of 128 cols; wave tile 32x64).
// Weights are pre-padded so the K-loop is branch-free; the true class count C
// is applied only in the bias load and the -inf mask of the reduction.
// ---------------------------------------------------------------------------
__global__ void __launch_bounds__(256) logits_lse_kernel(
    const __bf16* __restrict__ H, const __bf16* __restrict__ W,
    const float* __restrict__ Bv, const int* __restrict__ T,
    float* __restrict__ mP, float* __restrict__ sP, float* __restrict__ TL,
    int K, int C, int L) {
  __shared__ float lds_m[128][2];
  __shared__ float lds_s[128][2];
  __shared__ int   lds_t[128];

  const float NEG_INF = -__builtin_inff();
  int tid  = threadIdx.x;
  int lane = tid & 31;
  int wave = tid >> 5;
  int rg   = wave >> 1;   // 0..3 row group
  int cgp  = wave & 1;    // 0..1 col group
  int rowBlk = blockIdx.y * 128;
  int row0   = rowBlk + rg * 32;
  int cbase  = blockIdx.x * 512;

  if (tid < 128) lds_t[tid] = T[rowBlk + tid] - L;
  __syncthreads();

  int cg = lane >> 4, cl = lane & 15;
  int tloc[2][8];
#pragma unroll
  for (int tr = 0; tr < 2; ++tr)
#pragma unroll
    for (int j = 0; j < 8; ++j)
      tloc[tr][j] = lds_t[rg * 32 + tr * 16 + j + 8 * cg];

  float mrun[2][8], srun[2][8];
#pragma unroll
  for (int tr = 0; tr < 2; ++tr)
#pragma unroll
    for (int j = 0; j < 8; ++j) { mrun[tr][j] = NEG_INF; srun[tr][j] = 0.0f; }

  for (int ch = 0; ch < 4; ++ch) {
    int c0 = cbase + ch * 128 + cgp * 64;   // wave-uniform, always within padded W

    v8f acc[2][4] = {};
    for (int k0 = 0; k0 < K; k0 += 32) {
      v16bf a0 = load_a_frag(H, K, row0,      k0, lane);
      v16bf a1 = load_a_frag(H, K, row0 + 16, k0, lane);
      v16bf b0 = load_b_frag(W, K, c0,      k0, lane);
      v16bf b1 = load_b_frag(W, K, c0 + 16, k0, lane);
      v16bf b2 = load_b_frag(W, K, c0 + 32, k0, lane);
      v16bf b3 = load_b_frag(W, K, c0 + 48, k0, lane);
      acc[0][0] = wmma_bf16(a0, b0, acc[0][0]);
      acc[0][1] = wmma_bf16(a0, b1, acc[0][1]);
      acc[0][2] = wmma_bf16(a0, b2, acc[0][2]);
      acc[0][3] = wmma_bf16(a0, b3, acc[0][3]);
      acc[1][0] = wmma_bf16(a1, b0, acc[1][0]);
      acc[1][1] = wmma_bf16(a1, b1, acc[1][1]);
      acc[1][2] = wmma_bf16(a1, b2, acc[1][2]);
      acc[1][3] = wmma_bf16(a1, b3, acc[1][3]);
    }

    float bias[4];
#pragma unroll
    for (int tc = 0; tc < 4; ++tc)
      bias[tc] = (c0 + tc * 16 + cl < C) ? Bv[c0 + tc * 16 + cl] : 0.0f;

#pragma unroll
    for (int tr = 0; tr < 2; ++tr)
#pragma unroll
      for (int tc = 0; tc < 4; ++tc) {
        int col = c0 + tc * 16 + cl;
        bool valid = col < C;
#pragma unroll
        for (int j = 0; j < 8; ++j) {
          float v = acc[tr][tc][j] + bias[tc];
          if (valid && tloc[tr][j] == col)
            TL[rowBlk + rg * 32 + tr * 16 + j + 8 * cg] = v;  // exactly one writer/row
          float vv = valid ? v : NEG_INF;
          float M = fmaxf(mrun[tr][j], vv);
          if (M > NEG_INF) {
            srun[tr][j] = srun[tr][j] * __expf(mrun[tr][j] - M) + __expf(vv - M);
            mrun[tr][j] = M;
          }
        }
      }
  }

  // merge the 16 column-lanes of each half-wave (a row's 16 cols live there)
#pragma unroll
  for (int tr = 0; tr < 2; ++tr)
#pragma unroll
    for (int j = 0; j < 8; ++j) {
      float m = mrun[tr][j], s = srun[tr][j];
#pragma unroll
      for (int d = 1; d < 16; d <<= 1) {
        float mo = __shfl_xor(m, d, 32);
        float so = __shfl_xor(s, d, 32);
        float M = fmaxf(m, mo);
        if (M > NEG_INF) { s = s * __expf(m - M) + so * __expf(mo - M); m = M; }
        else             { s = 0.0f; }
      }
      mrun[tr][j] = m; srun[tr][j] = s;
    }

  if (cl == 0) {
#pragma unroll
    for (int tr = 0; tr < 2; ++tr)
#pragma unroll
      for (int j = 0; j < 8; ++j) {
        int rl = rg * 32 + tr * 16 + j + 8 * cg;
        lds_m[rl][cgp] = mrun[tr][j];
        lds_s[rl][cgp] = srun[tr][j];
      }
  }
  __syncthreads();
  if (tid < 128) {
    float m0 = lds_m[tid][0], s0 = lds_s[tid][0];
    float m1 = lds_m[tid][1], s1 = lds_s[tid][1];
    float M = fmaxf(m0, m1);
    float S = 0.0f;
    if (M > NEG_INF) S = s0 * __expf(m0 - M) + s1 * __expf(m1 - M);
    size_t idx = (size_t)blockIdx.x * N_TOKENS + rowBlk + tid;
    mP[idx] = M; sP[idx] = S;
  }
}

// ---------------------------------------------------------------------------
// Kernel 3: per-token combine of class-block partials -> nll
// ---------------------------------------------------------------------------
__global__ void combine_kernel(const float* __restrict__ mP, const float* __restrict__ sP,
                               const float* __restrict__ TL, const int* __restrict__ T,
                               float* __restrict__ outNll) {
  int n = blockIdx.x * blockDim.x + threadIdx.x;
  if (n >= N_TOKENS) return;
  const float NEG_INF = -__builtin_inff();
  int t  = T[n];
  int ci = (t < 10000) ? 0 : (t < 30000) ? 1 : 2;
  const int nCB[3]  = {20, 40, 40};
  const int base[3] = {0, 20, 60};
  float M = NEG_INF, S = 0.0f;
  int b = base[ci], k = nCB[ci];
  for (int i = 0; i < k; ++i) {
    float m = mP[(size_t)(b + i) * N_TOKENS + n];
    float s = sP[(size_t)(b + i) * N_TOKENS + n];
    float Mn = fmaxf(M, m);
    if (Mn > NEG_INF) { S = S * __expf(M - Mn) + s * __expf(m - Mn); M = Mn; }
  }
  float lse = M + __logf(S);
  outNll[n] = lse - TL[(size_t)ci * N_TOKENS + n];
}

// ---------------------------------------------------------------------------
// Kernel 4: loss = sum(nll)
// ---------------------------------------------------------------------------
__global__ void loss_kernel(const float* __restrict__ nll, float* __restrict__ loss) {
  __shared__ float red[256];
  float acc = 0.0f;
  for (int i = threadIdx.x; i < N_TOKENS; i += 256) acc += nll[i];
  red[threadIdx.x] = acc;
  __syncthreads();
  for (int s = 128; s > 0; s >>= 1) {
    if (threadIdx.x < s) red[threadIdx.x] += red[threadIdx.x + s];
    __syncthreads();
  }
  if (threadIdx.x == 0) loss[0] = red[0];
}

// ---------------------------------------------------------------------------
extern "C" void kernel_launch(void* const* d_in, const int* in_sizes, int n_in,
                              void* d_out, int out_size, void* d_ws, size_t ws_size,
                              hipStream_t stream) {
  (void)in_sizes; (void)n_in; (void)out_size; (void)ws_size;
  const float* x   = (const float*)d_in[0];
  const int*   tgt = (const int*)d_in[1];
  const float* P[3]  = {(const float*)d_in[2], (const float*)d_in[5], (const float*)d_in[8]};
  const float* Wm[3] = {(const float*)d_in[3], (const float*)d_in[6], (const float*)d_in[9]};
  const float* Bv[3] = {(const float*)d_in[4], (const float*)d_in[7], (const float*)d_in[10]};

  const int PD[3]     = {1024, 512, 256};
  const int Cc[3]     = {10000, 20000, 20257};
  const int Cpad[3]   = {10240, 20480, 20480};  // nCB * 512
  const int Lc[3]     = {0, 10000, 30000};
  const int nCB[3]    = {20, 40, 40};           // ceil(C/512)
  const int cbBase[3] = {0, 20, 60};

  // ---- workspace layout (bytes, all 256B-aligned) ----
  char* base = (char*)d_ws;
  __bf16* xbf  = (__bf16*)(base);                         // 8192*1024       (16.78 MB)
  __bf16* hbf  = (__bf16*)(base + 16777216);              // 8192*1024       (16.78 MB)
  __bf16* pbf[3] = {
      (__bf16*)(base + 33554432),                         // 1024*1024
      (__bf16*)(base + 35651584),                         // 512*1024
      (__bf16*)(base + 36700160)};                        // 256*1024
  __bf16* wbf[3] = {
      (__bf16*)(base + 37224448),                         // 10240*1024
      (__bf16*)(base + 58195968),                         // 20480*512
      (__bf16*)(base + 79167488)};                        // 20480*256
  float* mP = (float*)(base + 89653248);                  // 100*8192
  float* sP = (float*)(base + 92930048);                  // 100*8192
  float* TL = (float*)(base + 96206848);                  // 3*8192

  float* out    = (float*)d_out;
  float* outNll = out + 1;

  // ---- stage 0: convert (and pad) everything to bf16 ----
  const int CVT_BLK = 2048;
  cvt_pad_kernel<<<CVT_BLK, 256, 0, stream>>>(x, xbf,
      (long)N_TOKENS * IN_DIM, (long)N_TOKENS * IN_DIM);
  for (int i = 0; i < 3; ++i) {
    cvt_pad_kernel<<<CVT_BLK, 256, 0, stream>>>(P[i], pbf[i],
        (long)PD[i] * IN_DIM, (long)PD[i] * IN_DIM);
    cvt_pad_kernel<<<CVT_BLK, 256, 0, stream>>>(Wm[i], wbf[i],
        (long)Cc[i] * PD[i], (long)Cpad[i] * PD[i]);
  }

  // ---- stages 1-2 per cluster ----
  for (int i = 0; i < 3; ++i) {
    dim3 gp(PD[i] / 128, N_TOKENS / 128);
    proj_gemm_kernel<<<gp, 256, 0, stream>>>(xbf, pbf[i], hbf, IN_DIM, PD[i]);
    dim3 gl(nCB[i], N_TOKENS / 128);
    logits_lse_kernel<<<gl, 256, 0, stream>>>(
        hbf, wbf[i], Bv[i], tgt,
        mP + (size_t)cbBase[i] * N_TOKENS, sP + (size_t)cbBase[i] * N_TOKENS,
        TL + (size_t)i * N_TOKENS, PD[i], Cc[i], Lc[i]);
  }
  combine_kernel<<<dim3(N_TOKENS / 256), 256, 0, stream>>>(mP, sP, TL, tgt, outNll);
  loss_kernel<<<1, 256, 0, stream>>>(outNll, out);
}
